// CustomDeepseekDBOMoE_31628139168123
// MI455X (gfx1250) — matrix operand order repaired and checked by reading, not compiled, not run on hardware
//
#include <hip/hip_runtime.h>
#include <hip/hip_bf16.h>

// ---------------- problem constants ----------------
#define T_DIM   1024
#define H_DIM   1024
#define E_NUM   64
#define KI      512      // routed intermediate, also shared-expert half
#define SI_DIM  1024
#define NG      8
#define GSZ     (E_NUM / NG)
#define TKG     3
#define TOPK    6
#define RSF     2.5f

#define DW_PER_FRAG   256            // 32x16 bf16 fragment = 1KB = 256 dwords
#define FRAG_EXP_DW   (1024 * 256)   // fragments per routed expert (1024) in dwords

typedef __attribute__((ext_vector_type(16))) __bf16 v16bf;
typedef __attribute__((ext_vector_type(8)))  float  v8f;

// fp32 -> bf16 bits, round-to-nearest-even
__device__ __forceinline__ unsigned bfbits(float f) {
    unsigned u = __builtin_bit_cast(unsigned, f);
    return (u + 0x7FFFu + ((u >> 16) & 1u)) >> 16;
}
__device__ __forceinline__ unsigned bfpack(float lo, float hi) {
    return bfbits(lo) | (bfbits(hi) << 16);
}

__device__ __forceinline__ v8f wmma_bf16(v16bf a, v16bf b, v8f c) {
    return __builtin_amdgcn_wmma_f32_16x16x32_bf16(
        false, a, false, b, (short)0, c, false, false);
}

__device__ __forceinline__ float silu(float x) {
    return x * (1.0f / (1.0f + __expf(-x)));
}

// ---- fragment-order accessors -------------------------------------------
// A-fragment (16x32, MxK): lane<16 -> M=lane, K in {0..7,16..23}; lane>=16 -> +8.
// Stored lane-major: fragment base + lane*8 dwords (32B contiguous per lane).
union Frag16 { uint4 q[2]; v16bf v; };

__device__ __forceinline__ v16bf load_a_frag(const unsigned* U, int kt, int lane) {
    const uint4* p = (const uint4*)(U + kt * DW_PER_FRAG + lane * 8);
    Frag16 t;
    t.q[0] = p[0];
    t.q[1] = p[1];
    return t.v;
}

__device__ __forceinline__ v16bf load_b_frag(const unsigned* __restrict__ base,
                                             int fragIdx, int lane) {
    const uint4* p = (const uint4*)(base + (size_t)fragIdx * DW_PER_FRAG + lane * 8);
    Frag16 t;
    t.q[0] = p[0];
    t.q[1] = p[1];
    return t.v;
}

// element (m, kk-within-32) -> position inside A fragment
__device__ __forceinline__ void store_x_pair(unsigned* U, int row, int col, unsigned pk) {
    int kt = col >> 5, kk = col & 31;               // col even
    int laneT = row + 16 * ((kk >> 3) & 1);
    int j = ((kk >> 1) & 3) + 4 * (kk >> 4);
    U[kt * DW_PER_FRAG + laneT * 8 + j] = pk;
}
__device__ __forceinline__ void store_act_bf16(unsigned* U, int row, int col, float v) {
    int kt = col >> 5, kk = col & 31;
    int laneT = row + 16 * ((kk >> 3) & 1);
    int j = ((kk >> 1) & 3) + 4 * (kk >> 4);
    ((unsigned short*)U)[(kt * DW_PER_FRAG + laneT * 8 + j) * 2 + (kk & 1)] =
        (unsigned short)bfbits(v);
}

// ---------------- weight pre-conversion: fp32 [nMat][KDIM][NDIM] ->
// bf16 B-fragments, fragIdx = mat*FPM + nt*KT + kt, lane-major inside fragment.
template <int KDIM, int NDIM>
__global__ __launch_bounds__(256)
void convert_b_kernel(const float* __restrict__ src, unsigned* __restrict__ dst,
                      int nMat) {
    constexpr int KT  = KDIM / 32;
    constexpr int NT  = NDIM / 16;
    constexpr int FPM = KT * NT;

    long idx  = (long)blockIdx.x * 256 + threadIdx.x;  // one 32B lane-chunk each
    int  lane = (int)(idx & 31);
    long frag = idx >> 5;
    int  mat  = (int)(frag / FPM);
    if (mat >= nMat) return;
    int fi = (int)(frag % FPM);
    int nt = fi / KT;
    int kt = fi % KT;

    int n     = nt * 16 + (lane & 15);
    int kbase = kt * 32 + (lane >> 4) * 8;
    const float* s = src + ((size_t)mat * KDIM + kbase) * NDIM + n;

    unsigned o[8];
#pragma unroll
    for (int j = 0; j < 8; ++j) {
        const float* q = s + (size_t)((j & 3) * 2 + (j >> 2) * 16) * NDIM;
        o[j] = bfpack(q[0], q[NDIM]);
    }
    uint4* d = (uint4*)(dst + frag * DW_PER_FRAG + lane * 8);
    uint4 q0; q0.x = o[0]; q0.y = o[1]; q0.z = o[2]; q0.w = o[3];
    uint4 q1; q1.x = o[4]; q1.y = o[5]; q1.z = o[6]; q1.w = o[7];
    d[0] = q0;
    d[1] = q1;
}

// ---------------- kernel 0: zero per-expert counters ----------------
__global__ void init_counts_kernel(int* __restrict__ cnt) {
    if (threadIdx.x < E_NUM) cnt[threadIdx.x] = 0;
}

// ---------------- kernel 1: routing (noaux_tc grouped top-k) ----------------
__global__ __launch_bounds__(E_NUM)
void routing_kernel(const float* __restrict__ x,
                    const float* __restrict__ gate_w,
                    const float* __restrict__ e_bias,
                    int*   __restrict__ cnt,
                    int*   __restrict__ tokL,
                    float* __restrict__ wtL) {
    const int t = blockIdx.x;
    const int e = threadIdx.x;

    float acc = 0.0f;
    const float* xr = x + (size_t)t * H_DIM;
    for (int h = 0; h < H_DIM; ++h)
        acc += xr[h] * gate_w[(size_t)h * E_NUM + e];

    const float score = 1.0f / (1.0f + __expf(-acc));

    __shared__ float s_score[E_NUM];
    __shared__ float s_sc[E_NUM];
    __shared__ float s_g[NG];
    s_score[e] = score;
    s_sc[e]    = score + e_bias[e];
    __syncthreads();

    if (e < NG) {
        float m1 = -3.4e38f, m2 = -3.4e38f;
        for (int j = 0; j < GSZ; ++j) {
            float v = s_sc[e * GSZ + j];
            if (v > m1) { m2 = m1; m1 = v; }
            else if (v > m2) { m2 = v; }
        }
        s_g[e] = m1 + m2;
    }
    __syncthreads();

    if (e == 0) {
        bool gsel[NG];
        for (int g = 0; g < NG; ++g) gsel[g] = false;
        for (int it = 0; it < TKG; ++it) {
            float best = -3.4e38f; int bi = 0;
            for (int g = 0; g < NG; ++g)
                if (!gsel[g] && s_g[g] > best) { best = s_g[g]; bi = g; }
            gsel[bi] = true;
        }
        for (int g = 0; g < NG; ++g)
            if (!gsel[g])
                for (int j = 0; j < GSZ; ++j) s_sc[g * GSZ + j] = -3.4e38f;

        int   ids[TOPK];
        float ws[TOPK];
        float wsum = 0.0f;
        for (int it = 0; it < TOPK; ++it) {
            float best = -3.4e38f; int bi = 0;
            for (int i = 0; i < E_NUM; ++i)
                if (s_sc[i] > best) { best = s_sc[i]; bi = i; }
            s_sc[bi] = -3.4e38f;
            ids[it] = bi;
            ws[it]  = s_score[bi];
            wsum   += ws[it];
        }
        const float inv = 1.0f / wsum;
        for (int it = 0; it < TOPK; ++it) {
            int be  = ids[it];
            int pos = atomicAdd(&cnt[be], 1);
            tokL[(size_t)be * T_DIM + pos] = t;
            wtL [(size_t)be * T_DIM + pos] = ws[it] * inv;
        }
    }
}

// ---------------- 16-token FFN tile via WMMA (bf16 fragments) ----------------
template <bool SHARED, bool STORE>
__global__ __launch_bounds__(128)
void ffn_tile_kernel(const float*    __restrict__ x,
                     const unsigned* __restrict__ wgB,   // pre-offset frag bases
                     const unsigned* __restrict__ wuB,
                     const unsigned* __restrict__ wdB,
                     int KTd,                            // kt count in wd layout
                     const int*   __restrict__ cnt,
                     const int*   __restrict__ tokL,
                     const float* __restrict__ wtL,
                     float* __restrict__ out) {
    __shared__ unsigned XsU[16 * H_DIM / 2];  // 32 KB: X tile, A-fragment order
    __shared__ unsigned AsU[16 * KI / 2];     // 16 KB: act tile, A-fragment order

    const int tile = blockIdx.x;
    const int e    = SHARED ? 0 : blockIdx.y;

    int nTok;
    const int*   myTok = nullptr;
    const float* myWt  = nullptr;
    const unsigned *wg, *wu, *wd;
    if constexpr (SHARED) {
        nTok = 16;
        wg = wgB; wu = wuB; wd = wdB;
    } else {
        nTok = cnt[e] - tile * 16;
        if (nTok <= 0) return;                // uniform over the block
        if (nTok > 16) nTok = 16;
        myTok = tokL + (size_t)e * T_DIM + tile * 16;
        myWt  = wtL  + (size_t)e * T_DIM + tile * 16;
        wg = wgB + (size_t)e * FRAG_EXP_DW;
        wu = wuB + (size_t)e * FRAG_EXP_DW;
        wd = wdB + (size_t)e * FRAG_EXP_DW;
    }

    const int tid  = threadIdx.x;
    const int lane = tid & 31;
    const int wave = tid >> 5;

    // ---- gather 16 token rows, fp32 -> bf16 fragments in LDS ----
    {
        const int row   = tid >> 3;           // 8 threads per row
        const int cbase = (tid & 7) * 128;
        int token;
        if constexpr (SHARED) token = tile * 16 + row;
        else                  token = (row < nTok) ? myTok[row] : 0;
        const float4* src = (const float4*)(x + (size_t)token * H_DIM + cbase);
#pragma unroll 8
        for (int j = 0; j < 32; ++j) {
            float4 v = src[j];
            store_x_pair(XsU, row, cbase + 4 * j,     bfpack(v.x, v.y));
            store_x_pair(XsU, row, cbase + 4 * j + 2, bfpack(v.z, v.w));
        }
    }
    __syncthreads();

    // ---- phase 1: gate & up projections + SiLU -> act fragments in LDS ----
    for (int i0 = wave * 32; i0 < KI; i0 += 128) {
        const int nt0 = i0 >> 4;              // two N-tiles: nt0, nt0+1
        v8f cg0 = {0,0,0,0,0,0,0,0}, cg1 = {0,0,0,0,0,0,0,0};
        v8f cu0 = {0,0,0,0,0,0,0,0}, cu1 = {0,0,0,0,0,0,0,0};
        for (int kt = 0; kt < H_DIM / 32; ++kt) {
            v16bf a   = load_a_frag(XsU, kt, lane);
            v16bf bg0 = load_b_frag(wg, (nt0    ) * 32 + kt, lane);
            v16bf bg1 = load_b_frag(wg, (nt0 + 1) * 32 + kt, lane);
            v16bf bu0 = load_b_frag(wu, (nt0    ) * 32 + kt, lane);
            v16bf bu1 = load_b_frag(wu, (nt0 + 1) * 32 + kt, lane);
            cg0 = wmma_bf16(a, bg0, cg0);
            cg1 = wmma_bf16(a, bg1, cg1);
            cu0 = wmma_bf16(a, bu0, cu0);
            cu1 = wmma_bf16(a, bu1, cu1);
        }
        const int n  = lane & 15;
        const int kh = lane >> 4;
#pragma unroll
        for (int r = 0; r < 8; ++r) {
            const int row = r + 8 * kh;
            store_act_bf16(AsU, row, i0 + n,      silu(cg0[r]) * cu0[r]);
            store_act_bf16(AsU, row, i0 + 16 + n, silu(cg1[r]) * cu1[r]);
        }
    }
    __syncthreads();

    // ---- phase 2: down projection, scale, combine ----
    for (int c = 0; c < 4; ++c) {
        const int nb = wave * 256 + c * 64;
        const int nt = nb >> 4;
        v8f o0 = {0,0,0,0,0,0,0,0}, o1 = {0,0,0,0,0,0,0,0};
        v8f o2 = {0,0,0,0,0,0,0,0}, o3 = {0,0,0,0,0,0,0,0};
        for (int kt = 0; kt < KI / 32; ++kt) {
            v16bf a  = load_a_frag(AsU, kt, lane);
            v16bf b0 = load_b_frag(wd, (nt + 0) * KTd + kt, lane);
            v16bf b1 = load_b_frag(wd, (nt + 1) * KTd + kt, lane);
            v16bf b2 = load_b_frag(wd, (nt + 2) * KTd + kt, lane);
            v16bf b3 = load_b_frag(wd, (nt + 3) * KTd + kt, lane);
            o0 = wmma_bf16(a, b0, o0);
            o1 = wmma_bf16(a, b1, o1);
            o2 = wmma_bf16(a, b2, o2);
            o3 = wmma_bf16(a, b3, o3);
        }
        const int n  = lane & 15;
        const int kh = lane >> 4;
#pragma unroll
        for (int r = 0; r < 8; ++r) {
            const int row = r + 8 * kh;
            if constexpr (SHARED) {
                const int token = tile * 16 + row;
                float* dst = out + (size_t)token * H_DIM + nb + n;
                if constexpr (STORE) {
                    dst[0]  = o0[r];
                    dst[16] = o1[r];
                    dst[32] = o2[r];
                    dst[48] = o3[r];
                } else {
                    atomicAdd(dst + 0,  o0[r]);
                    atomicAdd(dst + 16, o1[r]);
                    atomicAdd(dst + 32, o2[r]);
                    atomicAdd(dst + 48, o3[r]);
                }
            } else {
                if (row < nTok) {
                    const int   token = myTok[row];
                    const float w     = myWt[row] * RSF;
                    float* dst = out + (size_t)token * H_DIM + nb + n;
                    atomicAdd(dst + 0,  o0[r] * w);
                    atomicAdd(dst + 16, o1[r] * w);
                    atomicAdd(dst + 32, o2[r] * w);
                    atomicAdd(dst + 48, o3[r] * w);
                }
            }
        }
    }
}

// ---------------- host launcher ----------------
extern "C" void kernel_launch(void* const* d_in, const int* in_sizes, int n_in,
                              void* d_out, int out_size, void* d_ws, size_t ws_size,
                              hipStream_t stream) {
    (void)in_sizes; (void)n_in; (void)out_size; (void)ws_size;

    const float* x       = (const float*)d_in[0];
    const float* gate_w  = (const float*)d_in[1];
    const float* e_bias  = (const float*)d_in[2];
    const float* w_gate  = (const float*)d_in[3];
    const float* w_up    = (const float*)d_in[4];
    const float* w_down  = (const float*)d_in[5];
    const float* sw_gate = (const float*)d_in[6];
    const float* sw_up   = (const float*)d_in[7];
    const float* sw_down = (const float*)d_in[8];
    float* out = (float*)d_out;

    char* ws = (char*)d_ws;
    size_t off = 0;
    int*   cnt  = (int*)(ws + off);   off += 1024;
    int*   tokL = (int*)(ws + off);   off += (size_t)E_NUM * T_DIM * 4;
    float* wtL  = (float*)(ws + off); off += (size_t)E_NUM * T_DIM * 4;

    const size_t szGU = (size_t)E_NUM * H_DIM * KI * 2;     // 64 MB each
    const size_t szSH = (size_t)H_DIM * SI_DIM * 2;         // 2 MB each
    unsigned* wgBf  = (unsigned*)(ws + off); off += szGU;
    unsigned* wuBf  = (unsigned*)(ws + off); off += szGU;
    unsigned* wdBf  = (unsigned*)(ws + off); off += szGU;
    unsigned* swgBf = (unsigned*)(ws + off); off += szSH;
    unsigned* swuBf = (unsigned*)(ws + off); off += szSH;
    unsigned* swdBf = (unsigned*)(ws + off); off += szSH;

    // routing
    init_counts_kernel<<<1, 64, 0, stream>>>(cnt);
    routing_kernel<<<T_DIM, E_NUM, 0, stream>>>(x, gate_w, e_bias, cnt, tokL, wtL);

    // weight pre-conversion into WMMA B-fragment order
    {
        const int blkGU = (E_NUM * (H_DIM / 32) * (KI / 16) * 32) / 256;   // 8192
        const int blkD  = (E_NUM * (KI / 32) * (H_DIM / 16) * 32) / 256;   // 8192
        const int blkS  = ((SI_DIM / 16) * (H_DIM / 32) * 32) / 256;       // 256
        convert_b_kernel<H_DIM, KI><<<blkGU, 256, 0, stream>>>(w_gate, wgBf, E_NUM);
        convert_b_kernel<H_DIM, KI><<<blkGU, 256, 0, stream>>>(w_up,   wuBf, E_NUM);
        convert_b_kernel<KI, H_DIM><<<blkD,  256, 0, stream>>>(w_down, wdBf, E_NUM);
        convert_b_kernel<H_DIM, SI_DIM><<<blkS, 256, 0, stream>>>(sw_gate, swgBf, 1);
        convert_b_kernel<H_DIM, SI_DIM><<<blkS, 256, 0, stream>>>(sw_up,   swuBf, 1);
        convert_b_kernel<SI_DIM, H_DIM><<<blkS, 256, 0, stream>>>(sw_down, swdBf, 1);
    }

    // shared expert: SI=1024 as two K=512 halves (write, then atomic-add).
    // swg/swu half1: skip 32 N-tiles => 32 * (H/32=32) * 256 dwords.
    // swd  half1: skip 16 K-tiles inside nt-major layout => 16 * 256 dwords.
    const size_t swguHalf = (size_t)(SI_DIM / 16 / 2) * (H_DIM / 32) * DW_PER_FRAG;
    const size_t swdHalf  = (size_t)(SI_DIM / 32 / 2) * DW_PER_FRAG;
    ffn_tile_kernel<true, true><<<dim3(T_DIM / 16, 1), 128, 0, stream>>>(
        x, swgBf, swuBf, swdBf, SI_DIM / 32,
        nullptr, nullptr, nullptr, out);
    ffn_tile_kernel<true, false><<<dim3(T_DIM / 16, 1), 128, 0, stream>>>(
        x, swgBf + swguHalf, swuBf + swguHalf, swdBf + swdHalf, SI_DIM / 32,
        nullptr, nullptr, nullptr, out);

    // routed experts
    ffn_tile_kernel<false, false><<<dim3(T_DIM / 16, E_NUM), 128, 0, stream>>>(
        x, wgBf, wuBf, wdBf, KI / 32,
        cnt, tokL, wtL, out);
}
